// LinearMessagePassingLayer_Residual_62749472195029
// MI455X (gfx1250) — matrix operand order repaired and checked by reading, not compiled, not run on hardware
//
#include <hip/hip_runtime.h>
#include <hip/hip_bf16.h>

typedef __attribute__((ext_vector_type(16))) __bf16 v16bf;
typedef __attribute__((ext_vector_type(8)))  float  v8f;

#define N_NODES 50000
#define N_EDGES 400000
#define D_NODE  256
#define D_EDGE  64
#define K_MSG   (D_NODE + D_EDGE)   // 320
#define LN_EPS  1e-6f

// ---------------------------------------------------------------------------
// Workspace prep
// ---------------------------------------------------------------------------
__global__ void zero_f32_kernel(float* __restrict__ p, long n) {
    long i = (long)blockIdx.x * blockDim.x + threadIdx.x;
    if (i < n) p[i] = 0.0f;
}

// Wt[n][k] = bf16(W[k][n])   (W is [K][N] row-major as in x @ W)
__global__ void transpose_bf16_kernel(const float* __restrict__ W,
                                      __bf16* __restrict__ Wt, int K, int N) {
    long i = (long)blockIdx.x * blockDim.x + threadIdx.x;
    if (i < (long)K * N) {
        int n = (int)(i / K);
        int k = (int)(i % K);
        Wt[i] = (__bf16)W[(long)k * N + n];
    }
}

// ---------------------------------------------------------------------------
// Edge message GEMM + scatter-add into agg
//   messages = [nodes[senders] | edges] @ W_message ; agg[recv] += messages
// Block: 512 thr = 16 waves; 16 edges per block; wave w owns output cols 16w..
// ---------------------------------------------------------------------------
__global__ __launch_bounds__(512) void edge_msg_kernel(
    const float* __restrict__ nodes, const float* __restrict__ edges,
    const int* __restrict__ senders, const int* __restrict__ receivers,
    const __bf16* __restrict__ WtMsg,   // [256][320] bf16, N-major
    float* __restrict__ agg)
{
    __shared__ __bf16 Abf[16][K_MSG];   // 16 edges x 320 K, bf16 (10 KB)
    __shared__ int    recv[16];

    const int ebase = blockIdx.x * 16;
    const int t = threadIdx.x;

    // Stage A operand (gather senders, concat edge feats, cvt to bf16)
    for (int idx = t; idx < 16 * K_MSG; idx += 512) {
        int e = idx / K_MSG, k = idx - e * K_MSG;
        int ge = ebase + e;
        float v;
        if (k < D_NODE) v = nodes[(long)senders[ge] * D_NODE + k];
        else            v = edges[(long)ge * D_EDGE + (k - D_NODE)];
        Abf[e][k] = (__bf16)v;
    }
    if (t < 16) recv[t] = receivers[ebase + t];
    __syncthreads();

    const int wave = t >> 5;           // 0..15 == N-tile
    const int lane = t & 31;
    const int half = lane >> 4;        // 0 | 1
    const int l16  = lane & 15;

    v8f acc = {};
    // B rows are contiguous: Wt row n, 2 runs of 8 consecutive k (kb+half*8, kb+16+half*8) per lane
    const __bf16* __restrict__ wrow = WtMsg + (long)(wave * 16 + l16) * K_MSG;

    for (int kb = 0; kb < K_MSG; kb += 32) {
        v16bf a, b;
        // A layout: lane -> row m=l16; elems 0..7 -> k = kb+half*8+j,
        //           elems 8..15 -> k = kb+16+half*8+j
        const __bf16* ap0 = &Abf[l16][kb + half * 8];
        const __bf16* ap1 = &Abf[l16][kb + 16 + half * 8];
#pragma unroll
        for (int j = 0; j < 8; ++j) { a[j] = ap0[j]; a[8 + j] = ap1[j]; }
        // B layout: lane -> col n=l16; elems 0..7 -> k = kb+half*8+j,
        //           elems 8..15 -> k = kb+16+half*8+j
        const __bf16* bp = wrow + kb + half * 8;
#pragma unroll
        for (int j = 0; j < 8; ++j) { b[j] = bp[j]; b[8 + j] = bp[16 + j]; }
        acc = __builtin_amdgcn_wmma_f32_16x16x32_bf16(
            false, a, false, b, (short)0, acc, false, false);
    }

    // D layout: elem v -> row m = v + half*8, col n = wave*16 + l16
    const int n = wave * 16 + l16;
#pragma unroll
    for (int v = 0; v < 8; ++v) {
        int m = v + half * 8;
        atomicAdd(&agg[(long)recv[m] * D_NODE + n], acc[v]);
    }
}

// ---------------------------------------------------------------------------
// Node update: h = LN(relu(agg)+nodes); 3-layer residual MLP (WMMA);
//              out = LN(nodes + x3)
// Block: 512 thr = 16 waves; 16 nodes per block; wave w owns output cols 16w..
// ---------------------------------------------------------------------------
__global__ __launch_bounds__(512) void node_update_kernel(
    const float* __restrict__ nodes, const float* __restrict__ agg,
    const __bf16* __restrict__ Wt0, const __bf16* __restrict__ Wt1,
    const __bf16* __restrict__ Wt2,
    const float* __restrict__ b0, const float* __restrict__ b1,
    const float* __restrict__ b2,
    const float* __restrict__ lnAs, const float* __restrict__ lnAb,
    const float* __restrict__ lnOs, const float* __restrict__ lnOb,
    float* __restrict__ out)
{
    __shared__ float  xf[16][D_NODE];   // fp32 running residual (16 KB)
    __shared__ __bf16 hb[16][D_NODE];   // bf16 shadow for WMMA A (8 KB)

    const int nbase = blockIdx.x * 16;
    const int t = threadIdx.x;
    const int wave = t >> 5;
    const int lane = t & 31;
    const int half = lane >> 4;
    const int l16  = lane & 15;

    // relu(agg) + nodes
    for (int idx = t; idx < 16 * D_NODE; idx += 512) {
        int r = idx >> 8, c = idx & 255;
        float a = agg[(long)(nbase + r) * D_NODE + c];
        a = a > 0.0f ? a : 0.0f;
        xf[r][c] = a + nodes[(long)(nbase + r) * D_NODE + c];
    }
    __syncthreads();

    // LayerNorm (aggr): wave w normalizes row w
    {
        float s = 0.0f, s2 = 0.0f;
        for (int c = lane; c < D_NODE; c += 32) {
            float v = xf[wave][c]; s += v; s2 += v * v;
        }
#pragma unroll
        for (int off = 16; off > 0; off >>= 1) {
            s  += __shfl_down(s,  off, 32);
            s2 += __shfl_down(s2, off, 32);
        }
        s = __shfl(s, 0, 32); s2 = __shfl(s2, 0, 32);
        float mu  = s * (1.0f / D_NODE);
        float var = s2 * (1.0f / D_NODE) - mu * mu;
        float inv = rsqrtf(var + LN_EPS);
        for (int c = lane; c < D_NODE; c += 32) {
            float h = (xf[wave][c] - mu) * inv * lnAs[c] + lnAb[c];
            xf[wave][c] = h;
            hb[wave][c] = (__bf16)h;
        }
    }
    __syncthreads();

    // 3-layer residual MLP; each layer one 16x16 tile per wave, K=256
    for (int l = 0; l < 3; ++l) {
        const __bf16* __restrict__ Wt = (l == 0) ? Wt0 : ((l == 1) ? Wt1 : Wt2);
        const float*  __restrict__ bb = (l == 0) ? b0  : ((l == 1) ? b1  : b2);

        v8f acc = {};
        const __bf16* __restrict__ wrow = Wt + (long)(wave * 16 + l16) * D_NODE;
        for (int kb = 0; kb < D_NODE; kb += 32) {
            v16bf a, b;
            const __bf16* ap0 = &hb[l16][kb + half * 8];
            const __bf16* ap1 = &hb[l16][kb + 16 + half * 8];
#pragma unroll
            for (int j = 0; j < 8; ++j) { a[j] = ap0[j]; a[8 + j] = ap1[j]; }
            const __bf16* bp = wrow + kb + half * 8;
#pragma unroll
            for (int j = 0; j < 8; ++j) { b[j] = bp[j]; b[8 + j] = bp[16 + j]; }
            acc = __builtin_amdgcn_wmma_f32_16x16x32_bf16(
                false, a, false, b, (short)0, acc, false, false);
        }
        __syncthreads();   // all waves done reading hb before we mutate it

        const int n = wave * 16 + l16;
        float bias = bb[n];
#pragma unroll
        for (int v = 0; v < 8; ++v) {
            int m = v + half * 8;
            float y = acc[v] + bias;
            if (l < 2) y = y > 0.0f ? y : 0.0f;   // ReLU on all but last
            float xn = xf[m][n] + y;              // residual (widths match)
            xf[m][n] = xn;
            hb[m][n] = (__bf16)xn;
        }
        __syncthreads();
    }

    // Final: out = LN(nodes + x3)
    {
        float s = 0.0f, s2 = 0.0f;
        for (int c = lane; c < D_NODE; c += 32) {
            float v = xf[wave][c] + nodes[(long)(nbase + wave) * D_NODE + c];
            xf[wave][c] = v; s += v; s2 += v * v;
        }
#pragma unroll
        for (int off = 16; off > 0; off >>= 1) {
            s  += __shfl_down(s,  off, 32);
            s2 += __shfl_down(s2, off, 32);
        }
        s = __shfl(s, 0, 32); s2 = __shfl(s2, 0, 32);
        float mu  = s * (1.0f / D_NODE);
        float var = s2 * (1.0f / D_NODE) - mu * mu;
        float inv = rsqrtf(var + LN_EPS);
        for (int c = lane; c < D_NODE; c += 32)
            out[(long)(nbase + wave) * D_NODE + c] =
                (xf[wave][c] - mu) * inv * lnOs[c] + lnOb[c];
    }
}

// ---------------------------------------------------------------------------
extern "C" void kernel_launch(void* const* d_in, const int* in_sizes, int n_in,
                              void* d_out, int out_size, void* d_ws, size_t ws_size,
                              hipStream_t stream) {
    (void)in_sizes; (void)n_in; (void)out_size; (void)ws_size;

    const float* nodes     = (const float*)d_in[0];
    const float* edges     = (const float*)d_in[1];
    const int*   senders   = (const int*)d_in[2];
    const int*   receivers = (const int*)d_in[3];
    const float* W_message = (const float*)d_in[4];
    const float* mlp_W0    = (const float*)d_in[5];
    const float* mlp_b0    = (const float*)d_in[6];
    const float* mlp_W1    = (const float*)d_in[7];
    const float* mlp_b1    = (const float*)d_in[8];
    const float* mlp_W2    = (const float*)d_in[9];
    const float* mlp_b2    = (const float*)d_in[10];
    const float* lnAs      = (const float*)d_in[11];
    const float* lnAb      = (const float*)d_in[12];
    const float* lnOs      = (const float*)d_in[13];
    const float* lnOb      = (const float*)d_in[14];
    float* out = (float*)d_out;

    // Workspace layout: agg fp32 [N,256] | WtMsg bf16 [256][320] | 3x Wt bf16 [256][256]
    float* agg = (float*)d_ws;
    char*  wsb = (char*)d_ws + (size_t)N_NODES * D_NODE * sizeof(float);
    __bf16* WtMsg = (__bf16*)wsb;
    __bf16* Wt0   = WtMsg + (size_t)256 * K_MSG;
    __bf16* Wt1   = Wt0 + (size_t)256 * 256;
    __bf16* Wt2   = Wt1 + (size_t)256 * 256;

    long aggN = (long)N_NODES * D_NODE;
    zero_f32_kernel<<<(unsigned)((aggN + 255) / 256), 256, 0, stream>>>(agg, aggN);

    transpose_bf16_kernel<<<(K_MSG * 256 + 255) / 256, 256, 0, stream>>>(
        W_message, WtMsg, K_MSG, 256);
    transpose_bf16_kernel<<<(256 * 256 + 255) / 256, 256, 0, stream>>>(
        mlp_W0, Wt0, 256, 256);
    transpose_bf16_kernel<<<(256 * 256 + 255) / 256, 256, 0, stream>>>(
        mlp_W1, Wt1, 256, 256);
    transpose_bf16_kernel<<<(256 * 256 + 255) / 256, 256, 0, stream>>>(
        mlp_W2, Wt2, 256, 256);

    edge_msg_kernel<<<N_EDGES / 16, 512, 0, stream>>>(
        nodes, edges, senders, receivers, WtMsg, agg);

    node_update_kernel<<<N_NODES / 16, 512, 0, stream>>>(
        nodes, agg, Wt0, Wt1, Wt2, mlp_b0, mlp_b1, mlp_b2,
        lnAs, lnAb, lnOs, lnOb, out);
}
